// InfiniteContextAttention_46780783788220
// MI455X (gfx1250) — matrix operand order repaired and checked by reading, not compiled, not run on hardware
//
#include <hip/hip_runtime.h>

// ---------------- problem dims (fixed by the reference) ----------------
#define EMBED   2048
#define LATTICE 8192
#define TOKENS  4096          // BATCH * SEQ = 2 * 2048

// ---------------- tiling ----------------
#define BM 128                // token rows per workgroup tile
#define BN 128                // lattice/embed cols per workgroup tile
#define BK 32                 // K slab = one bf16 WMMA K-step
#define LDST (BK + 8)         // padded LDS row stride (elements) -> 80B rows
#define NTHREADS 256          // 8 waves (wave32)

typedef __bf16 bf16_t;
typedef __attribute__((ext_vector_type(16))) __bf16 v16bf;
typedef __attribute__((ext_vector_type(8)))  __bf16 v8bf;
typedef __attribute__((ext_vector_type(4)))  __bf16 v4bf;
typedef __attribute__((ext_vector_type(8)))  float  v8f;

// ---------------- async global->LDS path (gfx1250), guarded ----------------
#if defined(__has_builtin)
#if __has_builtin(__builtin_amdgcn_global_load_async_to_lds_b128)
#define HAVE_ASYNC_CP 1
#endif
#if __has_builtin(__builtin_amdgcn_s_wait_asynccnt)
#define HAVE_ASYNC_WAIT 1
#endif
#endif
#if defined(HAVE_ASYNC_CP) && defined(HAVE_ASYNC_WAIT)
#define USE_ASYNC 1
#endif

#if defined(USE_ASYNC)
// The builtin expects pointers to a 16-byte int vector in AS1 (global) and
// AS3 (LDS): clang prints these as "int __vector_size__(16) __device__/__shared__ *".
typedef int v4i __attribute__((vector_size(16)));
typedef __attribute__((address_space(1))) v4i* gv4i_p;
typedef __attribute__((address_space(3))) v4i* lv4i_p;

__device__ __forceinline__ gv4i_p to_as1(const void* p) {
  // global generic pointers are numerically identical to AS1 pointers
  return (gv4i_p)(unsigned long long)p;
}
__device__ __forceinline__ lv4i_p to_as3(void* p) {
  // generic pointers into LDS carry the LDS byte offset in the low 32 bits
  return (lv4i_p)(unsigned int)(unsigned long long)p;
}
#endif

__device__ __forceinline__ void cp_b128(const bf16_t* g, bf16_t* l) {
#if defined(USE_ASYNC)
  __builtin_amdgcn_global_load_async_to_lds_b128(to_as1(g), to_as3(l), 0, 0);
#else
  *(int4*)l = *(const int4*)g;   // synchronous fallback: load + ds_store
#endif
}

__device__ __forceinline__ void stage_commit() {
#if defined(USE_ASYNC)
  __builtin_amdgcn_s_wait_asynccnt(0);
#endif
  __syncthreads();
}

// ---------------- helpers ----------------
__device__ __forceinline__ bf16_t f2bf(float f) {
  unsigned u = __builtin_bit_cast(unsigned, f);
  u += 0x7fffu + ((u >> 16) & 1u);          // round-to-nearest-even
  unsigned short h = (unsigned short)(u >> 16);
  return __builtin_bit_cast(bf16_t, h);
}

// Copy one 128 x 32 bf16 tile (row-major src, leading dim ldg) into padded LDS.
// 8 KB per tile = 512 x 16B chunks; 256 threads -> 2 chunks each.
__device__ __forceinline__ void stage_tile(const bf16_t* __restrict__ g, int ldg,
                                           bf16_t* __restrict__ l, int tid) {
#pragma unroll
  for (int i = 0; i < 2; ++i) {
    int c   = tid + i * NTHREADS;
    int row = c >> 2;              // 4 chunks per 64B row
    int col = (c & 3) << 3;        // element offset (8 bf16 = 16B)
    cp_b128(g + (size_t)row * ldg + col, l + row * LDST + col);
  }
}

// A fragment, 16x32 bf16: lanes 0-15 hold row m, K {0..7,16..23};
// lanes 16-31 hold K {8..15,24..31}.  rowp points at LDS row m.
__device__ __forceinline__ v16bf frag_a(const bf16_t* rowp, int lane) {
  int klo = (lane >> 4) << 3;                    // 0 or 8
  v8bf lo = *(const v8bf*)(rowp + klo);
  v8bf hi = *(const v8bf*)(rowp + klo + 16);
  return __builtin_shufflevector(lo, hi, 0,1,2,3,4,5,6,7,8,9,10,11,12,13,14,15);
}

// B fragment, 32x16 bf16: lane holds column n = lane%16; lanes 0-15 K {0..7,16..23},
// lanes 16-31 K {8..15,24..31}.
// Since weights are stored [N,K] row-major, a B column is a contiguous W row.
__device__ __forceinline__ v16bf frag_b(const bf16_t* rowp, int lane) {
  int klo = (lane >> 4) << 3;                    // 0 or 8
  v8bf lo = *(const v8bf*)(rowp + klo);
  v8bf hi = *(const v8bf*)(rowp + klo + 16);
  return __builtin_shufflevector(lo, hi, 0,1,2,3,4,5,6,7,8,9,10,11,12,13,14,15);
}

// ---------------- f32 -> bf16 conversion (memory-bound, vectorized) ----------------
__global__ __launch_bounds__(NTHREADS)
void cvt_bf16_kernel(const float* __restrict__ src, bf16_t* __restrict__ dst, int n4) {
  int i      = blockIdx.x * blockDim.x + threadIdx.x;
  int stride = gridDim.x * blockDim.x;
  const float4* s4 = (const float4*)src;
  v4bf* d4 = (v4bf*)dst;
  for (; i < n4; i += stride) {
    float4 f = s4[i];
    v4bf o;
    o[0] = f2bf(f.x); o[1] = f2bf(f.y); o[2] = f2bf(f.z); o[3] = f2bf(f.w);
    d4[i] = o;
  }
}

// ---------------- fused QV GEMM: attn = (X*Wq^T) .* (X*Wv^T), bf16 out ----------------
__global__ __launch_bounds__(NTHREADS)
void qv_gemm_kernel(const bf16_t* __restrict__ X,    // [TOKENS, EMBED]
                    const bf16_t* __restrict__ Wq,   // [LATTICE, EMBED]
                    const bf16_t* __restrict__ Wv,   // [LATTICE, EMBED]
                    bf16_t* __restrict__ Attn) {     // [TOKENS, LATTICE]
  __shared__ __align__(16) bf16_t sA[2][BM * LDST];
  __shared__ __align__(16) bf16_t sQ[2][BN * LDST];
  __shared__ __align__(16) bf16_t sV[2][BN * LDST];

  const int tid  = threadIdx.x;
  const int lane = tid & 31;
  const int wave = tid >> 5;
  const int wm   = wave >> 1;            // 0..3 : 32-row wave slice
  const int wn   = wave & 1;             // 0..1 : 64-col wave slice
  const int blockM = blockIdx.y * BM;
  const int blockN = blockIdx.x * BN;

  v8f accq[2][4], accv[2][4];
#pragma unroll
  for (int mt = 0; mt < 2; ++mt)
#pragma unroll
    for (int nt = 0; nt < 4; ++nt) {
      accq[mt][nt] = (v8f){0.f,0.f,0.f,0.f,0.f,0.f,0.f,0.f};
      accv[mt][nt] = (v8f){0.f,0.f,0.f,0.f,0.f,0.f,0.f,0.f};
    }

  const int nks = EMBED / BK;            // 64 K-steps

  stage_tile(X  + (size_t)blockM * EMBED, EMBED, sA[0], tid);
  stage_tile(Wq + (size_t)blockN * EMBED, EMBED, sQ[0], tid);
  stage_tile(Wv + (size_t)blockN * EMBED, EMBED, sV[0], tid);
  stage_commit();

  for (int ks = 0; ks < nks; ++ks) {
    const int cur = ks & 1;
    if (ks + 1 < nks) {                  // prefetch next slab into other buffer
      const int kb = (ks + 1) * BK;
      stage_tile(X  + (size_t)blockM * EMBED + kb, EMBED, sA[cur ^ 1], tid);
      stage_tile(Wq + (size_t)blockN * EMBED + kb, EMBED, sQ[cur ^ 1], tid);
      stage_tile(Wv + (size_t)blockN * EMBED + kb, EMBED, sV[cur ^ 1], tid);
    }

    v16bf af[2];
#pragma unroll
    for (int mt = 0; mt < 2; ++mt) {
      int m = wm * 32 + mt * 16 + (lane & 15);
      af[mt] = frag_a(&sA[cur][m * LDST], lane);
    }
#pragma unroll
    for (int nt = 0; nt < 4; ++nt) {
      int n = wn * 64 + nt * 16 + (lane & 15);
      v16bf bq = frag_b(&sQ[cur][n * LDST], lane);
      v16bf bv = frag_b(&sV[cur][n * LDST], lane);
#pragma unroll
      for (int mt = 0; mt < 2; ++mt) {
        accq[mt][nt] = __builtin_amdgcn_wmma_f32_16x16x32_bf16(
            false, af[mt], false, bq, (short)0, accq[mt][nt], false, false);
        accv[mt][nt] = __builtin_amdgcn_wmma_f32_16x16x32_bf16(
            false, af[mt], false, bv, (short)0, accv[mt][nt], false, false);
      }
    }
    stage_commit();
  }

  // epilogue: attn = q .* v, cast to bf16.  C layout: lane n=lane%16,
  // VGPR r -> row (lane/16)*8 + r.
#pragma unroll
  for (int mt = 0; mt < 2; ++mt)
#pragma unroll
    for (int nt = 0; nt < 4; ++nt) {
      int col   = blockN + wn * 64 + nt * 16 + (lane & 15);
      int rbase = blockM + wm * 32 + mt * 16 + ((lane >> 4) << 3);
#pragma unroll
      for (int r = 0; r < 8; ++r) {
        float p = accq[mt][nt][r] * accv[mt][nt][r];
        Attn[(size_t)(rbase + r) * LATTICE + col] = f2bf(p);
      }
    }
}

// ---------------- output GEMM: out = attn * Wout^T, f32 out ----------------
__global__ __launch_bounds__(NTHREADS)
void out_gemm_kernel(const bf16_t* __restrict__ A,   // [TOKENS, LATTICE]
                     const bf16_t* __restrict__ B,   // [EMBED, LATTICE]
                     float* __restrict__ C) {        // [TOKENS, EMBED]
  __shared__ __align__(16) bf16_t sA[2][BM * LDST];
  __shared__ __align__(16) bf16_t sB[2][BN * LDST];

  const int tid  = threadIdx.x;
  const int lane = tid & 31;
  const int wave = tid >> 5;
  const int wm   = wave >> 1;
  const int wn   = wave & 1;
  const int blockM = blockIdx.y * BM;
  const int blockN = blockIdx.x * BN;

  v8f acc[2][4];
#pragma unroll
  for (int mt = 0; mt < 2; ++mt)
#pragma unroll
    for (int nt = 0; nt < 4; ++nt)
      acc[mt][nt] = (v8f){0.f,0.f,0.f,0.f,0.f,0.f,0.f,0.f};

  const int nks = LATTICE / BK;          // 256 K-steps

  stage_tile(A + (size_t)blockM * LATTICE, LATTICE, sA[0], tid);
  stage_tile(B + (size_t)blockN * LATTICE, LATTICE, sB[0], tid);
  stage_commit();

  for (int ks = 0; ks < nks; ++ks) {
    const int cur = ks & 1;
    if (ks + 1 < nks) {
      const int kb = (ks + 1) * BK;
      stage_tile(A + (size_t)blockM * LATTICE + kb, LATTICE, sA[cur ^ 1], tid);
      stage_tile(B + (size_t)blockN * LATTICE + kb, LATTICE, sB[cur ^ 1], tid);
    }

    v16bf af[2];
#pragma unroll
    for (int mt = 0; mt < 2; ++mt) {
      int m = wm * 32 + mt * 16 + (lane & 15);
      af[mt] = frag_a(&sA[cur][m * LDST], lane);
    }
#pragma unroll
    for (int nt = 0; nt < 4; ++nt) {
      int n = wn * 64 + nt * 16 + (lane & 15);
      v16bf bf = frag_b(&sB[cur][n * LDST], lane);
#pragma unroll
      for (int mt = 0; mt < 2; ++mt)
        acc[mt][nt] = __builtin_amdgcn_wmma_f32_16x16x32_bf16(
            false, af[mt], false, bf, (short)0, acc[mt][nt], false, false);
    }
    stage_commit();
  }

#pragma unroll
  for (int mt = 0; mt < 2; ++mt)
#pragma unroll
    for (int nt = 0; nt < 4; ++nt) {
      int col   = blockN + wn * 64 + nt * 16 + (lane & 15);
      int rbase = blockM + wm * 32 + mt * 16 + ((lane >> 4) << 3);
#pragma unroll
      for (int r = 0; r < 8; ++r)
        C[(size_t)(rbase + r) * EMBED + col] = acc[mt][nt][r];
    }
}

// ---------------- host launcher ----------------
extern "C" void kernel_launch(void* const* d_in, const int* in_sizes, int n_in,
                              void* d_out, int out_size, void* d_ws, size_t ws_size,
                              hipStream_t stream) {
  (void)in_sizes; (void)n_in; (void)out_size; (void)ws_size;

  const float* x  = (const float*)d_in[0];   // hidden_states [4096, 2048]
  const float* wq = (const float*)d_in[1];   // Wq [8192, 2048]
  // d_in[2] = Wk is dead in the reference -> skipped entirely (saves 137 GFLOP)
  const float* wv = (const float*)d_in[3];   // Wv [8192, 2048]
  const float* wo = (const float*)d_in[4];   // Wout [2048, 8192]

  char* ws = (char*)d_ws;
  bf16_t* xb   = (bf16_t*)ws;  ws += (size_t)TOKENS  * EMBED   * sizeof(bf16_t);
  bf16_t* wqb  = (bf16_t*)ws;  ws += (size_t)LATTICE * EMBED   * sizeof(bf16_t);
  bf16_t* wvb  = (bf16_t*)ws;  ws += (size_t)LATTICE * EMBED   * sizeof(bf16_t);
  bf16_t* wob  = (bf16_t*)ws;  ws += (size_t)EMBED   * LATTICE * sizeof(bf16_t);
  bf16_t* attn = (bf16_t*)ws;  // [TOKENS, LATTICE] bf16

  // f32 -> bf16 (one-time, memory-bound)
  cvt_bf16_kernel<<<2048, NTHREADS, 0, stream>>>(x,  xb,  TOKENS  * EMBED   / 4);
  cvt_bf16_kernel<<<2048, NTHREADS, 0, stream>>>(wq, wqb, LATTICE * EMBED   / 4);
  cvt_bf16_kernel<<<2048, NTHREADS, 0, stream>>>(wv, wvb, LATTICE * EMBED   / 4);
  cvt_bf16_kernel<<<2048, NTHREADS, 0, stream>>>(wo, wob, EMBED   * LATTICE / 4);

  // fused q/v projection + elementwise product
  dim3 g1(LATTICE / BN, TOKENS / BM);    // (64, 32)
  qv_gemm_kernel<<<g1, NTHREADS, 0, stream>>>(xb, wqb, wvb, attn);

  // output projection
  dim3 g2(EMBED / BN, TOKENS / BM);      // (16, 32)
  out_gemm_kernel<<<g2, NTHREADS, 0, stream>>>(attn, wob, (float*)d_out);
}
